// TemporalDecoder_25074019074456
// MI455X (gfx1250) — matrix operand order repaired and checked by reading, not compiled, not run on hardware
//
#include <hip/hip_runtime.h>

// TemporalDecoder: 2-layer LSTM (B=256, H=512) encode S=128 steps, then 32
// autoregressive decode steps with FC [512x512] projection.
// bf16 WMMA (v_wmma_f32_16x16x32_bf16) with fp32 accumulation, software-
// pipelined K loop (prefetch next A + 4 B tiles before current WMMAs),
// fused fp32->bf16 input conversion in the layer-0 cell (no cvt kernels).

typedef unsigned short u16;
typedef unsigned int   u32;
typedef __attribute__((ext_vector_type(16))) __bf16 v16bf;
typedef __attribute__((ext_vector_type(8)))  float  v8f;
typedef __attribute__((ext_vector_type(4)))  u32    u32x4;
typedef __attribute__((ext_vector_type(4)))  float  f32x4;

struct U32x8 { u32x4 lo, hi; };

__device__ __forceinline__ u16 f2bf(float x) {
  u32 u = __builtin_bit_cast(u32, x);
  u32 r = u + 0x7FFFu + ((u >> 16) & 1u);   // round-to-nearest-even
  return (u16)(r >> 16);
}

// Pack two fp32 -> packed bf16 pair (f0 in low half) with RNE, via v_perm_b32.
__device__ __forceinline__ u32 pkbf2(float f0, float f1) {
  u32 a = __builtin_bit_cast(u32, f0);
  u32 b = __builtin_bit_cast(u32, f1);
  a += 0x7FFFu + ((a >> 16) & 1u);
  b += 0x7FFFu + ((b >> 16) & 1u);
  // combined bytes: src1(a)=bytes0-3, src0(b)=bytes4-7; pick a[3:2], b[3:2]
  return __builtin_amdgcn_perm(b, a, 0x07060302u);
}

__device__ __forceinline__ float sigf(float x) {
  return 1.0f / (1.0f + __expf(-x));
}

// A operand (16x32 bf16, MxK), ISA 7.12.2 layout:
//   lanes 0-15 : row M=lane,     K=k0+[0..7] (v0-3) and k0+[16..23] (v4-7)
//   lanes 16-31: row M=lane-16,  K=k0+[8..15]        and k0+[24..31]
__device__ __forceinline__ v16bf load_a16x32(const u16* __restrict__ base, int ld,
                                             int row0, int k0, int lane) {
  const u16* p = base + (row0 + (lane & 15)) * ld + k0 + ((lane >> 4) << 3);
  U32x8 u;
  u.lo = *reinterpret_cast<const u32x4*>(p);        // 8 bf16
  u.hi = *reinterpret_cast<const u32x4*>(p + 16);   // 8 bf16, +32B
  return __builtin_bit_cast(v16bf, u);
}

// Same A layout, but source is fp32 (strided slice); convert to bf16 in-reg.
__device__ __forceinline__ v16bf load_a16x32_f32(const float* __restrict__ src,
                                                 int rowStride, int colOff,
                                                 int row0, int k0, int lane) {
  const float* p = src + (row0 + (lane & 15)) * rowStride + colOff + k0 + ((lane >> 4) << 3);
  f32x4 f0 = *reinterpret_cast<const f32x4*>(p);
  f32x4 f1 = *reinterpret_cast<const f32x4*>(p + 4);
  f32x4 f2 = *reinterpret_cast<const f32x4*>(p + 16);
  f32x4 f3 = *reinterpret_cast<const f32x4*>(p + 20);
  U32x8 u;
  u.lo = (u32x4){ pkbf2(f0.x, f0.y), pkbf2(f0.z, f0.w),
                  pkbf2(f1.x, f1.y), pkbf2(f1.z, f1.w) };
  u.hi = (u32x4){ pkbf2(f2.x, f2.y), pkbf2(f2.z, f2.w),
                  pkbf2(f3.x, f3.y), pkbf2(f3.z, f3.w) };
  return __builtin_bit_cast(v16bf, u);
}

// B operand (32x16 bf16, KxN) fed from rows of B^T (row-major weight W[n][k]):
//   lane holds column n = col0+(lane&15); lanes 0-15: K=k0+[0..7],[16..23],
//   lanes 16-31: K=k0+[8..15],[24..31] (same K order as A).
__device__ __forceinline__ v16bf load_bT32x16(const u16* __restrict__ base, int ld,
                                              int col0, int k0, int lane) {
  const u16* p = base + (col0 + (lane & 15)) * ld + k0 + ((lane >> 4) << 3);
  U32x8 u;
  u.lo = *reinterpret_cast<const u32x4*>(p);
  u.hi = *reinterpret_cast<const u32x4*>(p + 16);
  return __builtin_bit_cast(v16bf, u);
}

__device__ __forceinline__ v8f wmma_bf16(v16bf a, v16bf b, v8f c) {
  return __builtin_amdgcn_wmma_f32_16x16x32_bf16(
      /*neg_a=*/false, a, /*neg_b=*/false, b,
      /*c_mod=*/(short)0, c, /*reuse_a=*/false, /*reuse_b=*/false);
}

// Unified A-tile fetch over the concatenated K=1024: first 512 = layer input
// (fp32 slice if F32IN, else bf16 [256][512]); last 512 = recurrent bf16 h.
template<bool F32IN>
__device__ __forceinline__ v16bf load_input_tile(const void* __restrict__ Ain,
                                                 int aRowStride, int aColOff,
                                                 const u16* __restrict__ Hprev,
                                                 int mbase, int kb, int lane) {
  if (kb < 16) {
    if (F32IN)
      return load_a16x32_f32((const float*)Ain, aRowStride, aColOff, mbase, kb * 32, lane);
    else
      return load_a16x32((const u16*)Ain, 512, mbase, kb * 32, lane);
  }
  return load_a16x32(Hprev, 512, mbase, (kb - 16) * 32, lane);
}

// ---------------------------------------------------------------------------
// Fused LSTM cell: gates = [Ain | Hprev] @ Wcat^T + bias, then pointwise.
// Wcat: [2048][1024] bf16 (rows = 4 gates x 512; cols 0..511 input weights,
// 512..1023 recurrent). cst: [256][512] fp32 (in place). Hnew: bf16 out.
// Grid: 64 blocks x 256 threads = 512 waves; one 16x16 (m,j) tile per wave,
// all 4 gate tiles computed per wave (A reused across 4 WMMAs).
// K loop software-pipelined: prefetch next A + 4 B before current WMMAs.
// ---------------------------------------------------------------------------
template<bool F32IN>
__global__ __launch_bounds__(256) void lstm_cell_wmma(
    const void* __restrict__ Ain, int aRowStride, int aColOff,
    const u16* __restrict__ Hprev,
    const u16* __restrict__ Wcat, const float* __restrict__ bias,
    float* __restrict__ cst, u16* __restrict__ Hnew)
{
  const int lane = threadIdx.x & 31;
  const int wid  = (blockIdx.x * blockDim.x + threadIdx.x) >> 5;   // 0..511
  const int jt = wid & 31;          // 32 column tiles (H=512)
  const int mt = wid >> 5;          // 16 row tiles (B=256)
  const int mbase = mt << 4, jbase = jt << 4;
  const int col = jbase + (lane & 15);

  v8f acc[4];
  #pragma unroll
  for (int g = 0; g < 4; ++g) {
    const float bv = bias[(g << 9) + col];
    v8f a;
    #pragma unroll
    for (int r = 0; r < 8; ++r) a[r] = bv;
    acc[g] = a;
  }

  // Pipeline prologue
  v16bf a_cur = load_input_tile<F32IN>(Ain, aRowStride, aColOff, Hprev, mbase, 0, lane);
  v16bf b_cur[4];
  #pragma unroll
  for (int g = 0; g < 4; ++g)
    b_cur[g] = load_bT32x16(Wcat + (g << 9) * 1024, 1024, jbase, 0, lane);

  for (int kb = 0; kb < 31; ++kb) {
    v16bf a_nxt = load_input_tile<F32IN>(Ain, aRowStride, aColOff, Hprev, mbase, kb + 1, lane);
    v16bf b_nxt[4];
    #pragma unroll
    for (int g = 0; g < 4; ++g)
      b_nxt[g] = load_bT32x16(Wcat + (g << 9) * 1024, 1024, jbase, (kb + 1) * 32, lane);
    #pragma unroll
    for (int g = 0; g < 4; ++g)
      acc[g] = wmma_bf16(a_cur, b_cur[g], acc[g]);
    a_cur = a_nxt;
    #pragma unroll
    for (int g = 0; g < 4; ++g) b_cur[g] = b_nxt[g];
  }
  #pragma unroll
  for (int g = 0; g < 4; ++g)
    acc[g] = wmma_bf16(a_cur, b_cur[g], acc[g]);

  // Pointwise. C/D layout: VGPR r, lanes 0-15: M=r, lanes 16-31: M=8+r
  const int rbase = mbase + ((lane >> 4) << 3);
  #pragma unroll
  for (int r = 0; r < 8; ++r) {
    const int idx = ((rbase + r) << 9) + col;
    const float gi = acc[0][r], gf = acc[1][r], gg = acc[2][r], go = acc[3][r];
    const float cp = cst[idx];
    const float cn = sigf(gf) * cp + sigf(gi) * tanhf(gg);
    const float h  = sigf(go) * tanhf(cn);
    cst[idx]  = cn;
    Hnew[idx] = f2bf(h);
  }
}

// ---------------------------------------------------------------------------
// FC: pred = Hin @ Wout^T + bout ; fp32 to out[b, step, 0, f]. Pipelined K.
// ---------------------------------------------------------------------------
__global__ __launch_bounds__(256) void fc_wmma(
    const u16* __restrict__ Hin, const u16* __restrict__ Wout,
    const float* __restrict__ bout, float* __restrict__ out, int step)
{
  const int lane = threadIdx.x & 31;
  const int wid  = (blockIdx.x * blockDim.x + threadIdx.x) >> 5;
  const int jt = wid & 31, mt = wid >> 5;
  const int mbase = mt << 4, jbase = jt << 4;
  const int col = jbase + (lane & 15);

  const float bv = bout[col];
  v8f acc;
  #pragma unroll
  for (int r = 0; r < 8; ++r) acc[r] = bv;

  v16bf a_cur = load_a16x32(Hin, 512, mbase, 0, lane);
  v16bf b_cur = load_bT32x16(Wout, 512, jbase, 0, lane);
  for (int kb = 0; kb < 15; ++kb) {
    v16bf a_nxt = load_a16x32(Hin, 512, mbase, (kb + 1) * 32, lane);
    v16bf b_nxt = load_bT32x16(Wout, 512, jbase, (kb + 1) * 32, lane);
    acc = wmma_bf16(a_cur, b_cur, acc);
    a_cur = a_nxt; b_cur = b_nxt;
  }
  acc = wmma_bf16(a_cur, b_cur, acc);

  const int rbase = mbase + ((lane >> 4) << 3);
  #pragma unroll
  for (int r = 0; r < 8; ++r)
    out[(rbase + r) * (32 * 512) + step * 512 + col] = acc[r];  // [B][32][1][F]
}

// Pack [Wih | Whh] -> Wcat bf16, 2 layers x [2048][1024]
__global__ void pack_wcat_kernel(u16* __restrict__ Wcat,
                                 const float* __restrict__ Wih,
                                 const float* __restrict__ Whh)
{
  const int idx = blockIdx.x * blockDim.x + threadIdx.x;   // 4194304
  const int l = idx >> 21, rem = idx & 0x1FFFFF;
  const int n = rem >> 10, k = rem & 1023;
  const float v = (k < 512) ? Wih[(l * 2048 + n) * 512 + k]
                            : Whh[(l * 2048 + n) * 512 + (k - 512)];
  Wcat[idx] = f2bf(v);
}

__global__ void pack_wout_kernel(u16* __restrict__ dst, const float* __restrict__ src)
{
  const int idx = blockIdx.x * blockDim.x + threadIdx.x;   // 262144
  dst[idx] = f2bf(src[idx]);
}

__global__ void bias_sum_kernel(float* __restrict__ bias,
                                const float* __restrict__ b_ih,
                                const float* __restrict__ b_hh)
{
  const int idx = blockIdx.x * blockDim.x + threadIdx.x;   // 4096
  bias[idx] = b_ih[idx] + b_hh[idx];
}

__global__ void zero_kernel(u32* __restrict__ p)
{
  p[blockIdx.x * blockDim.x + threadIdx.x] = 0u;
}

// ---------------------------------------------------------------------------
extern "C" void kernel_launch(void* const* d_in, const int* in_sizes, int n_in,
                              void* d_out, int out_size, void* d_ws, size_t ws_size,
                              hipStream_t stream)
{
  const float* x     = (const float*)d_in[0];   // [256][128][512]
  const float* W_ih  = (const float*)d_in[1];   // [2][2048][512]
  const float* W_hh  = (const float*)d_in[2];   // [2][2048][512]
  const float* b_ih  = (const float*)d_in[3];   // [2][2048]
  const float* b_hh  = (const float*)d_in[4];   // [2][2048]
  const float* W_out = (const float*)d_in[5];   // [512][512]
  const float* b_out = (const float*)d_in[6];   // [512]
  float* out = (float*)d_out;                   // [256][32][1][512]
  (void)in_sizes; (void)n_in; (void)out_size; (void)ws_size;

  char* w = (char*)d_ws;
  u16*   WCAT  = (u16*)(w + 0);              // 2 x 2048 x 1024 bf16 = 8 MB
  u16*   WOUTB = (u16*)(w + 8388608);        // 512 x 512 bf16      = 512 KB
  float* BIAS  = (float*)(w + 8912896);      // 2 x 2048 fp32       = 16 KB
  float* CST   = (float*)(w + 8929280);      // 2 x 256 x 512 fp32  = 1 MB
  u16*   HB    = (u16*)(w + 9977856);        // 2 layers x 2 parity x [256][512] bf16
  const int HBn = 256 * 512;                 // elements per hidden buffer
  const size_t WCAT_L = (size_t)2048 * 1024; // per-layer Wcat elements

  // ---- prep (every launch; deterministic, graph-capture safe) ----
  pack_wcat_kernel<<<16384, 256, 0, stream>>>(WCAT, W_ih, W_hh);
  pack_wout_kernel<<<1024, 256, 0, stream>>>(WOUTB, W_out);
  bias_sum_kernel<<<16, 256, 0, stream>>>(BIAS, b_ih, b_hh);
  // zero c-state + all 4 hidden buffers: 2097152 bytes = 524288 dwords
  zero_kernel<<<2048, 256, 0, stream>>>((u32*)CST);

  int p0 = 0, p1 = 0;   // parity: HB[layer*2 + p] holds current hidden state

  // ---- encode: 128 steps (cell0 reads fp32 x slice directly) ----
  for (int t = 0; t < 128; ++t) {
    lstm_cell_wmma<true><<<64, 256, 0, stream>>>(
        (const void*)x, 128 * 512, t * 512, HB + (0 * 2 + p0) * HBn,
        WCAT, BIAS, CST, HB + (0 * 2 + (p0 ^ 1)) * HBn);
    p0 ^= 1;
    lstm_cell_wmma<false><<<64, 256, 0, stream>>>(
        (const void*)(HB + (0 * 2 + p0) * HBn), 512, 0, HB + (2 + p1) * HBn,
        WCAT + WCAT_L, BIAS + 2048, CST + HBn, HB + (2 + (p1 ^ 1)) * HBn);
    p1 ^= 1;
  }

  // ---- decode: 32 steps (cell0 reads fp32 preds from d_out directly) ----
  for (int s = 0; s < 32; ++s) {
    fc_wmma<<<64, 256, 0, stream>>>(HB + (2 + p1) * HBn, WOUTB, b_out, out, s);
    lstm_cell_wmma<true><<<64, 256, 0, stream>>>(
        (const void*)out, 32 * 512, s * 512, HB + (0 * 2 + p0) * HBn,
        WCAT, BIAS, CST, HB + (0 * 2 + (p0 ^ 1)) * HBn);
    p0 ^= 1;
    lstm_cell_wmma<false><<<64, 256, 0, stream>>>(
        (const void*)(HB + (0 * 2 + p0) * HBn), 512, 0, HB + (2 + p1) * HBn,
        WCAT + WCAT_L, BIAS + 2048, CST + HBn, HB + (2 + (p1 ^ 1)) * HBn);
    p1 ^= 1;
  }
}
